// GMNnet_44049184588262
// MI455X (gfx1250) — compile-verified, matrix operand order checked by reading, not run on hardware
//
#include <hip/hip_runtime.h>
#include <hip/hip_bf16.h>

// ---------------------------------------------------------------------------
// GMN forward on MI455X (gfx1250, wave32, WMMA bf16 16x16x32, f32 accum).
// ~140 GFLOP of GEMM vs tens of MB mandatory traffic -> compute bound ->
// everything heavy goes through v_wmma_f32_16x16x32_bf16; edge gathers go
// memory->LDS via global_load_async_to_lds_b128 (ASYNCcnt path).
// ---------------------------------------------------------------------------

typedef __attribute__((ext_vector_type(16))) __bf16 v16bf;
typedef __attribute__((ext_vector_type(8)))  __bf16 v8bf;
typedef __attribute__((ext_vector_type(8)))  float  v8f;

union FragU { v16bf v; v8bf h[2]; };

__device__ __forceinline__ __bf16 f2bf(float x) {
    union { float f; unsigned u; } v; v.f = x;
    unsigned r = v.u + 0x7FFFu + ((v.u >> 16) & 1u);   // round-to-nearest-even
    unsigned short hs = (unsigned short)(r >> 16);
    return __builtin_bit_cast(__bf16, hs);
}

__device__ __forceinline__ v8f zero8() {
    v8f z;
#pragma unroll
    for (int i = 0; i < 8; ++i) z[i] = 0.0f;
    return z;
}

// Load a 16x32 bf16 operand fragment (A-layout / N-major B-layout).
// Memory: 16 rows (M or N index) x K, row-major with leading dim `ld`.
// Lane L: r = L&15 selects row, hf = L>>4 selects K groups {hf*8..hf*8+7} and
// {16+hf*8 .. 16+hf*8+7} -> two contiguous 16-byte loads per lane.
__device__ __forceinline__ v16bf load_frag(const __bf16* __restrict__ base, int ld) {
    const int lane = threadIdx.x & 31;
    const int r  = lane & 15;
    const int hf = lane >> 4;
    const __bf16* p = base + r * ld + hf * 8;
    FragU u;
    u.h[0] = *reinterpret_cast<const v8bf*>(p);
    u.h[1] = *reinterpret_cast<const v8bf*>(p + 16);
    return u.v;
}

__device__ __forceinline__ v8f wmma_bf16(v16bf a, v16bf b, v8f c) {
    return __builtin_amdgcn_wmma_f32_16x16x32_bf16(false, a, false, b,
                                                   (short)0, c, false, false);
}

// CDNA5 async gather: LDS[lds_off + ...] = MEM[addr .. addr+15], per lane.
// GV mode (64-bit vaddr, saddr = off). Tracked by ASYNCcnt.
__device__ __forceinline__ void async_gather_b128(unsigned lds_off,
                                                  const void* gaddr) {
    asm volatile("global_load_async_to_lds_b128 %0, %1, off"
                 :: "v"(lds_off), "v"((unsigned long long)(uintptr_t)gaddr)
                 : "memory");
}

__device__ __forceinline__ void wait_async0() {
    asm volatile("s_wait_asynccnt 0x0" ::: "memory");
}

// ---------------------------------------------------------------------------
// Prep kernels
// ---------------------------------------------------------------------------

__global__ void f32_to_bf16_kernel(const float* __restrict__ in,
                                   __bf16* __restrict__ out, int n) {
    int idx = blockIdx.x * 256 + threadIdx.x;
    if (idx < n) out[idx] = f2bf(in[idx]);
}

__global__ void embed_gather_kernel(const int* __restrict__ ids,
                                    const float* __restrict__ table,
                                    float* __restrict__ h_f32,
                                    __bf16* __restrict__ h_bf, int n) {
    int idx = blockIdx.x * 256 + threadIdx.x;
    if (idx < n * 128) {
        int row = idx >> 7, d = idx & 127;
        float v = table[ids[row] * 128 + d];
        h_f32[idx] = v;
        h_bf[idx]  = f2bf(v);
    }
}

__global__ void transpose_bf16_kernel(const __bf16* __restrict__ in,  // [n,128]
                                      __bf16* __restrict__ out,       // [128,n]
                                      int n) {
    int idx = blockIdx.x * 256 + threadIdx.x;
    if (idx < n * 128) {
        int row = idx >> 7, d = idx & 127;
        out[d * n + row] = in[idx];
    }
}

// ---------------------------------------------------------------------------
// Message passing: m = scatter_add(relu([x_i|x_j|ew] @ W_msg^T + b), dst)
// One block = 16 edges; 8 waves cover the 128 output cols (16 each).
// A-tile gathered straight into LDS via global_load_async_to_lds_b128.
// ---------------------------------------------------------------------------

__global__ void msg_pass_kernel(const __bf16* __restrict__ h_bf,       // [N,128]
                                const int* __restrict__ ei,            // [2,E]
                                const int* __restrict__ ea,            // [E]
                                const __bf16* __restrict__ Wmsg_bf,    // [128,384]
                                const __bf16* __restrict__ ee_bf,      // [20,128]
                                const float* __restrict__ b_msg,       // [128]
                                float* __restrict__ m_out,             // [N,128]
                                int E_) {
    __shared__ __bf16 Abuf[16 * 384];
    __shared__ int dstIdx[16];
    const int tile = blockIdx.x;
    const int tid  = threadIdx.x;

    if (tid < 16) dstIdx[tid] = ei[E_ + tile * 16 + tid];

    // 16 edges x 48 16-byte segments, direct memory->LDS (ASYNCcnt path)
    for (int idx = tid; idx < 16 * 48; idx += 256) {
        int row = idx / 48, seg = idx % 48;
        int e = tile * 16 + row;
        const __bf16* src;
        if (seg < 16) {                       // x_i  (target features)
            src = h_bf + ei[E_ + e] * 128 + seg * 8;
        } else if (seg < 32) {                // x_j  (source features)
            src = h_bf + ei[e] * 128 + (seg - 16) * 8;
        } else {                              // edge embedding
            src = ee_bf + ea[e] * 128 + (seg - 32) * 8;
        }
        unsigned lds_off = (unsigned)(uintptr_t)(&Abuf[row * 384 + seg * 8]);
        async_gather_b128(lds_off, src);
    }
    wait_async0();          // each wave drains its own async loads...
    __syncthreads();        // ...then the barrier makes all LDS writes visible

    const int lane = tid & 31, wave = tid >> 5;
    const int r = lane & 15, hf = lane >> 4;

    v8f acc = zero8();
#pragma unroll
    for (int kb = 0; kb < 12; ++kb) {
        v16bf a = load_frag(&Abuf[kb * 32], 384);
        v16bf b = load_frag(Wmsg_bf + (wave * 16) * 384 + kb * 32, 384);
        acc = wmma_bf16(a, b, acc);
    }

    const int col  = wave * 16 + r;
    const float bb = b_msg[col];
#pragma unroll
    for (int j = 0; j < 8; ++j) {
        float v = fmaxf(acc[j] + bb, 0.0f);
        int d = dstIdx[j + 8 * hf];
        atomicAdd(&m_out[d * 128 + col], v);
    }
}

// ---------------------------------------------------------------------------
// Flash cross-attention: u = xq - softmax_row(xq @ xk^T) @ xk
// One wave = 16 query rows. Streams key cols in steps of 32 with online
// softmax; P->bf16 via per-wave LDS tile; P@V accumulated with WMMA.
// ---------------------------------------------------------------------------

#define FLASH_WAVES 8

__global__ void flash_attn_kernel(const float*  __restrict__ xq_f32,   // [N,128]
                                  const __bf16* __restrict__ xq_bf,    // [N,128]
                                  const __bf16* __restrict__ xk_bf,    // [N,128]
                                  const __bf16* __restrict__ xkT_bf,   // [128,N]
                                  float* __restrict__ u_out,           // [N,128]
                                  int n_cols) {
    __shared__ __bf16 ldsP[FLASH_WAVES * 16 * 32];
    const int lane = threadIdx.x & 31;
    const int wave = threadIdx.x >> 5;
    const int row_base = (blockIdx.x * FLASH_WAVES + wave) * 16;
    const int r  = lane & 15;
    const int hf = lane >> 4;
    __bf16* myP = &ldsP[wave * 16 * 32];

    // Query fragments for K=128 (hoisted: reused for every key tile)
    v16bf a[4];
#pragma unroll
    for (int kk = 0; kk < 4; ++kk)
        a[kk] = load_frag(xq_bf + row_base * 128 + kk * 32, 128);

    v8f o[8];
#pragma unroll
    for (int t = 0; t < 8; ++t) o[t] = zero8();
    float mrow[8], lrow[8];
#pragma unroll
    for (int j = 0; j < 8; ++j) { mrow[j] = -1e30f; lrow[j] = 0.0f; }

    for (int cb = 0; cb < n_cols; cb += 32) {
        // S tile: 16 rows x 32 cols
        v8f s0 = zero8(), s1 = zero8();
#pragma unroll
        for (int kk = 0; kk < 4; ++kk) {
            v16bf b0 = load_frag(xk_bf + cb * 128 + kk * 32, 128);
            v16bf b1 = load_frag(xk_bf + (cb + 16) * 128 + kk * 32, 128);
            s0 = wmma_bf16(a[kk], b0, s0);
            s1 = wmma_bf16(a[kk], b1, s1);
        }

        // online softmax: row stats live across the 16 lanes of a half-group
        float scale[8], newm[8];
#pragma unroll
        for (int j = 0; j < 8; ++j) {
            float t = fmaxf(s0[j], s1[j]);
            t = fmaxf(t, __shfl_xor(t, 1));
            t = fmaxf(t, __shfl_xor(t, 2));
            t = fmaxf(t, __shfl_xor(t, 4));
            t = fmaxf(t, __shfl_xor(t, 8));
            float nm = fmaxf(mrow[j], t);
            scale[j] = __expf(mrow[j] - nm);
            newm[j]  = nm;
            mrow[j]  = nm;
        }
#pragma unroll
        for (int j = 0; j < 8; ++j) {
            float p0 = __expf(s0[j] - newm[j]);
            float p1 = __expf(s1[j] - newm[j]);
            s0[j] = p0; s1[j] = p1;
            float ps = p0 + p1;
            ps += __shfl_xor(ps, 1);
            ps += __shfl_xor(ps, 2);
            ps += __shfl_xor(ps, 4);
            ps += __shfl_xor(ps, 8);
            lrow[j] = lrow[j] * scale[j] + ps;
        }

        // P -> bf16 A-fragment via per-wave LDS tile (in-wave RAW: dscnt wait)
#pragma unroll
        for (int j = 0; j < 8; ++j) {
            int row = j + 8 * hf;
            myP[row * 32 + r]      = f2bf(s0[j]);
            myP[row * 32 + 16 + r] = f2bf(s1[j]);
        }
        asm volatile("s_wait_dscnt 0x0" ::: "memory");
        v16bf pa = load_frag(myP, 32);

        // rescale and accumulate P @ V over all 128 feature cols
#pragma unroll
        for (int t = 0; t < 8; ++t) {
#pragma unroll
            for (int j = 0; j < 8; ++j) o[t][j] *= scale[j];
            v16bf bv = load_frag(xkT_bf + (t * 16) * n_cols + cb, n_cols);
            o[t] = wmma_bf16(pa, bv, o[t]);
        }
    }

    // u = xq - o / l
#pragma unroll
    for (int t = 0; t < 8; ++t) {
#pragma unroll
        for (int j = 0; j < 8; ++j) {
            int row = row_base + j + 8 * hf;
            int col = t * 16 + r;
            float val = o[t][j] / lrow[j];
            u_out[row * 128 + col] = xq_f32[row * 128 + col] - val;
        }
    }
}

// ---------------------------------------------------------------------------
// Fused GRU cell per 16-node tile:
//   gi = [m|u] @ W_ih^T + b_ih ; gh = h @ W_hh^T + b_hh ; gate math.
// ---------------------------------------------------------------------------

__global__ void gru_kernel(const float*  __restrict__ m_in,    // [N,128]
                           const float*  __restrict__ u_in,    // [N,128]
                           const float*  __restrict__ h_f32,   // [N,128]
                           const __bf16* __restrict__ h_bf,    // [N,128]
                           const __bf16* __restrict__ Wih_bf,  // [384,256]
                           const float*  __restrict__ b_ih,    // [384]
                           const __bf16* __restrict__ Whh_bf,  // [384,128]
                           const float*  __restrict__ b_hh,    // [384]
                           float*  __restrict__ hn_f32,        // [N,128]
                           __bf16* __restrict__ hn_bf) {       // [N,128]
    __shared__ __bf16 Abuf[16 * 256];
    __shared__ float gi[16 * 384];
    __shared__ float gh[16 * 384];
    const int tile = blockIdx.x;
    const int tid  = threadIdx.x;

    for (int idx = tid; idx < 16 * 32; idx += 256) {
        int row = idx >> 5, seg = idx & 31;
        int node = tile * 16 + row;
        const float* src = (seg < 16) ? (m_in + node * 128 + seg * 8)
                                      : (u_in + node * 128 + (seg - 16) * 8);
        __bf16* dstp = &Abuf[row * 256 + seg * 8];
#pragma unroll
        for (int q = 0; q < 8; ++q) dstp[q] = f2bf(src[q]);
    }
    __syncthreads();

    const int lane = tid & 31, wave = tid >> 5;
    const int r = lane & 15, hf = lane >> 4;

    for (int t = wave; t < 24; t += 8) {          // gi: 24 col-tiles of 16
        v8f acc = zero8();
#pragma unroll
        for (int kb = 0; kb < 8; ++kb) {          // K = 256
            v16bf av = load_frag(&Abuf[kb * 32], 256);
            v16bf bv = load_frag(Wih_bf + (t * 16) * 256 + kb * 32, 256);
            acc = wmma_bf16(av, bv, acc);
        }
        int col = t * 16 + r;
        float bb = b_ih[col];
#pragma unroll
        for (int j = 0; j < 8; ++j)
            gi[(j + 8 * hf) * 384 + col] = acc[j] + bb;
    }
    for (int t = wave; t < 24; t += 8) {          // gh: K = 128
        v8f acc = zero8();
#pragma unroll
        for (int kb = 0; kb < 4; ++kb) {
            v16bf av = load_frag(h_bf + (tile * 16) * 128 + kb * 32, 128);
            v16bf bv = load_frag(Whh_bf + (t * 16) * 128 + kb * 32, 128);
            acc = wmma_bf16(av, bv, acc);
        }
        int col = t * 16 + r;
        float bb = b_hh[col];
#pragma unroll
        for (int j = 0; j < 8; ++j)
            gh[(j + 8 * hf) * 384 + col] = acc[j] + bb;
    }
    __syncthreads();

    for (int idx = tid; idx < 16 * 128; idx += 256) {
        int row = idx >> 7, col = idx & 127;
        int node = tile * 16 + row;
        float ir = gi[row * 384 + col];
        float iz = gi[row * 384 + 128 + col];
        float in_ = gi[row * 384 + 256 + col];
        float hr = gh[row * 384 + col];
        float hz = gh[row * 384 + 128 + col];
        float hn = gh[row * 384 + 256 + col];
        float hold = h_f32[node * 128 + col];
        float rr = 1.0f / (1.0f + __expf(-(ir + hr)));
        float zz = 1.0f / (1.0f + __expf(-(iz + hz)));
        float nn = tanhf(in_ + rr * hn);
        float hv = (1.0f - zz) * nn + zz * hold;
        hn_f32[node * 128 + col] = hv;
        hn_bf[node * 128 + col]  = f2bf(hv);
    }
}

// ---------------------------------------------------------------------------
// Global attention pool: out = sum_n softmax(sigmoid(x@gw^T+gb))_n * x[n]
// ---------------------------------------------------------------------------

__global__ void pool_kernel(const float* __restrict__ h,       // [N,128]
                            const float* __restrict__ gate_w,  // [128]
                            const float* __restrict__ gate_b,  // [1]
                            float* __restrict__ gates_ws,      // [N]
                            float* __restrict__ out,           // [128]
                            int n) {
    __shared__ float red[256];
    __shared__ float sMax, sSum;
    const int tid = threadIdx.x;

    float lmax = -1e30f;
    for (int node = tid; node < n; node += 256) {
        float acc = gate_b[0];
        for (int d = 0; d < 128; ++d) acc += h[node * 128 + d] * gate_w[d];
        float g = 1.0f / (1.0f + __expf(-acc));
        gates_ws[node] = g;
        lmax = fmaxf(lmax, g);
    }
    red[tid] = lmax; __syncthreads();
    for (int s = 128; s > 0; s >>= 1) {
        if (tid < s) red[tid] = fmaxf(red[tid], red[tid + s]);
        __syncthreads();
    }
    if (tid == 0) sMax = red[0];
    __syncthreads();

    float lsum = 0.0f;
    for (int node = tid; node < n; node += 256) lsum += __expf(gates_ws[node] - sMax);
    red[tid] = lsum; __syncthreads();
    for (int s = 128; s > 0; s >>= 1) {
        if (tid < s) red[tid] += red[tid + s];
        __syncthreads();
    }
    if (tid == 0) sSum = red[0];
    __syncthreads();

    if (tid < 128) {
        float acc = 0.0f;
        for (int node = 0; node < n; ++node)
            acc += __expf(gates_ws[node] - sMax) / sSum * h[node * 128 + tid];
        out[tid] = acc;
    }
}

// ---------------------------------------------------------------------------
// Host-side orchestration
// ---------------------------------------------------------------------------

extern "C" void kernel_launch(void* const* d_in, const int* in_sizes, int n_in,
                              void* d_out, int out_size, void* d_ws, size_t ws_size,
                              hipStream_t stream) {
    const int N = in_sizes[0];            // 6144
    const int E = in_sizes[2] / 2;        // 196608

    const int*   x1   = (const int*)d_in[0];
    const int*   x2   = (const int*)d_in[1];
    const int*   ei1  = (const int*)d_in[2];
    const int*   ei2  = (const int*)d_in[3];
    const int*   ea1  = (const int*)d_in[4];
    const int*   ea2  = (const int*)d_in[5];
    const float* embed_w  = (const float*)d_in[6];
    const float* ee_w     = (const float*)d_in[7];
    const float* W_msg    = (const float*)d_in[8];
    const float* b_msg    = (const float*)d_in[9];
    const float* W_ih     = (const float*)d_in[10];
    const float* b_ih     = (const float*)d_in[11];
    const float* W_hh     = (const float*)d_in[12];
    const float* b_hh     = (const float*)d_in[13];
    const float* gate_w   = (const float*)d_in[14];
    const float* gate_b   = (const float*)d_in[15];
    float* out = (float*)d_out;

    // workspace carve-up (256-byte aligned)
    size_t off = 0;
    auto carve = [&](size_t bytes) -> void* {
        void* p = (char*)d_ws + off;
        off = (off + bytes + 255) & ~(size_t)255;
        return p;
    };
    const size_t nfB  = (size_t)N * 128 * sizeof(float);
    const size_t nbB  = (size_t)N * 128 * sizeof(__bf16);

    float*  h1f[2] = { (float*)carve(nfB), (float*)carve(nfB) };
    float*  h2f[2] = { (float*)carve(nfB), (float*)carve(nfB) };
    __bf16* h1b[2] = { (__bf16*)carve(nbB), (__bf16*)carve(nbB) };
    __bf16* h2b[2] = { (__bf16*)carve(nbB), (__bf16*)carve(nbB) };
    __bf16* h1T = (__bf16*)carve(nbB);
    __bf16* h2T = (__bf16*)carve(nbB);
    float*  m1  = (float*)carve(nfB);
    float*  m2  = (float*)carve(nfB);
    float*  u1  = (float*)carve(nfB);
    float*  u2  = (float*)carve(nfB);
    __bf16* Wmsg_bf = (__bf16*)carve(128 * 384 * sizeof(__bf16));
    __bf16* Wih_bf  = (__bf16*)carve(384 * 256 * sizeof(__bf16));
    __bf16* Whh_bf  = (__bf16*)carve(384 * 128 * sizeof(__bf16));
    __bf16* ee_bf   = (__bf16*)carve(20 * 128 * sizeof(__bf16));
    float*  gates1  = (float*)carve((size_t)N * sizeof(float));
    float*  gates2  = (float*)carve((size_t)N * sizeof(float));

    // weight conversion to bf16
    f32_to_bf16_kernel<<<(128*384 + 255)/256, 256, 0, stream>>>(W_msg, Wmsg_bf, 128*384);
    f32_to_bf16_kernel<<<(384*256 + 255)/256, 256, 0, stream>>>(W_ih,  Wih_bf,  384*256);
    f32_to_bf16_kernel<<<(384*128 + 255)/256, 256, 0, stream>>>(W_hh,  Whh_bf,  384*128);
    f32_to_bf16_kernel<<<(20*128  + 255)/256, 256, 0, stream>>>(ee_w,  ee_bf,   20*128);

    // node embeddings
    const int gBlocks = (N * 128 + 255) / 256;
    embed_gather_kernel<<<gBlocks, 256, 0, stream>>>(x1, embed_w, h1f[0], h1b[0], N);
    embed_gather_kernel<<<gBlocks, 256, 0, stream>>>(x2, embed_w, h2f[0], h2b[0], N);

    int cur = 0;
    const int msgBlocks   = E / 16;                 // 12288
    const int flashBlocks = N / (16 * FLASH_WAVES); // 48
    const int gruBlocks   = N / 16;                 // 384

    for (int layer = 0; layer < 2; ++layer) {
        hipMemsetAsync(m1, 0, nfB, stream);
        hipMemsetAsync(m2, 0, nfB, stream);

        msg_pass_kernel<<<msgBlocks, 256, 0, stream>>>(h1b[cur], ei1, ea1, Wmsg_bf,
                                                       ee_bf, b_msg, m1, E);
        msg_pass_kernel<<<msgBlocks, 256, 0, stream>>>(h2b[cur], ei2, ea2, Wmsg_bf,
                                                       ee_bf, b_msg, m2, E);

        transpose_bf16_kernel<<<gBlocks, 256, 0, stream>>>(h1b[cur], h1T, N);
        transpose_bf16_kernel<<<gBlocks, 256, 0, stream>>>(h2b[cur], h2T, N);

        flash_attn_kernel<<<flashBlocks, 256, 0, stream>>>(h1f[cur], h1b[cur],
                                                           h2b[cur], h2T, u1, N);
        flash_attn_kernel<<<flashBlocks, 256, 0, stream>>>(h2f[cur], h2b[cur],
                                                           h1b[cur], h1T, u2, N);

        gru_kernel<<<gruBlocks, 256, 0, stream>>>(m1, u1, h1f[cur], h1b[cur],
                                                  Wih_bf, b_ih, Whh_bf, b_hh,
                                                  h1f[1 - cur], h1b[1 - cur]);
        gru_kernel<<<gruBlocks, 256, 0, stream>>>(m2, u2, h2f[cur], h2b[cur],
                                                  Wih_bf, b_ih, Whh_bf, b_hh,
                                                  h2f[1 - cur], h2b[1 - cur]);
        cur = 1 - cur;
    }

    pool_kernel<<<1, 256, 0, stream>>>(h1f[cur], gate_w, gate_b, gates1, out,       N);
    pool_kernel<<<1, 256, 0, stream>>>(h2f[cur], gate_w, gate_b, gates2, out + 128, N);
}
